// NanoGPT_8744553415035
// MI455X (gfx1250) — matrix operand order repaired and checked by reading, not compiled, not run on hardware
//
#include <hip/hip_runtime.h>
#include <hip/hip_bf16.h>
#include <math.h>

// Problem constants (match reference)
#define NB 4
#define NT 1024
#define ND 1024
#define NL 8
#define NH 16
#define NHD 64
#define NV 50257
#define MTOT (NB * NT)   // 4096 rows

typedef __attribute__((ext_vector_type(16))) __bf16 v16bf;
typedef __attribute__((ext_vector_type(8)))  float  v8f;
typedef __attribute__((ext_vector_type(4)))  unsigned int v4u;
typedef __attribute__((ext_vector_type(8)))  int v8i;
typedef __attribute__((ext_vector_type(4)))  int v4i;

__device__ __forceinline__ unsigned short f32_to_bf16_rne(float f) {
    union { float f; unsigned u; } a; a.f = f;
    unsigned u = a.u;
    u += 0x7FFFu + ((u >> 16) & 1u);   // round-to-nearest-even
    return (unsigned short)(u >> 16);
}

union Frag { v16bf v; unsigned u[8]; };

// WMMA A-fragment (16-bit, 16x32) per-VGPR K column, per CDNA5 ISA 7.12.2:
//   VGPR j<4 -> k = kb + 2j ;  j>=4 -> k = 16 + kb + 2(j-4);  kb = 8*(lane/16)
__device__ __forceinline__ int a_kcol(int j, int kb) {
    return ((j >> 2) * 16) + kb + 2 * (j & 3);
}
// 16-lane-group reductions (rows of a C-layout tile live in one half-wave)
__device__ __forceinline__ float group16_max(float x) {
#pragma unroll
    for (int m = 1; m <= 8; m <<= 1) x = fmaxf(x, __shfl_xor(x, m, 32));
    return x;
}
__device__ __forceinline__ float group16_sum(float x) {
#pragma unroll
    for (int m = 1; m <= 8; m <<= 1) x += __shfl_xor(x, m, 32);
    return x;
}

// ---------------------------------------------------------------------------
// Embedding: x[b,t,:] = wte[idx[b,t],:] + wpe[t,:]
// ---------------------------------------------------------------------------
__global__ __launch_bounds__(256)
void embed_kernel(const int* __restrict__ idx, const float* __restrict__ wte,
                  const float* __restrict__ wpe, float* __restrict__ x) {
    int i = blockIdx.x * 256 + threadIdx.x;        // over MTOT*ND
    int d = i & (ND - 1);
    int m = i >> 10;                                // row = b*T + t
    int t = m & (NT - 1);
    x[i] = wte[(size_t)idx[m] * ND + d] + wpe[t * ND + d];
}

// ---------------------------------------------------------------------------
// LayerNorm over D=1024, one block (256 threads) per row
// ---------------------------------------------------------------------------
__global__ __launch_bounds__(256)
void ln_kernel(const float* __restrict__ x, const float* __restrict__ g,
               const float* __restrict__ b, float* __restrict__ out) {
    __shared__ float red[256];
    const int row = blockIdx.x;
    const int tid = threadIdx.x;
    const float* xr = x + (size_t)row * ND;
    float v[4];
    float s = 0.f;
#pragma unroll
    for (int i = 0; i < 4; ++i) { v[i] = xr[tid + 256 * i]; s += v[i]; }
    red[tid] = s; __syncthreads();
#pragma unroll
    for (int off = 128; off > 0; off >>= 1) {
        if (tid < off) red[tid] += red[tid + off];
        __syncthreads();
    }
    float mu = red[0] * (1.0f / ND);
    __syncthreads();
    float s2 = 0.f;
#pragma unroll
    for (int i = 0; i < 4; ++i) { float d = v[i] - mu; s2 += d * d; }
    red[tid] = s2; __syncthreads();
#pragma unroll
    for (int off = 128; off > 0; off >>= 1) {
        if (tid < off) red[tid] += red[tid + off];
        __syncthreads();
    }
    float var = red[0] * (1.0f / ND);
    float r = rsqrtf(var + 1e-5f);
    float* orow = out + (size_t)row * ND;
#pragma unroll
    for (int i = 0; i < 4; ++i) {
        int c = tid + 256 * i;
        orow[c] = (v[i] - mu) * r * g[c] + b[c];
    }
}

// ---------------------------------------------------------------------------
// WMMA bf16 GEMM:  C = epilogue(A[M,K] @ B[K,N])   (see round-2 comments)
// Block tile 128x128x64, 8 waves (4Mx2N), wave 32x64 -> 16 v_wmma per stage.
// Epilogue staged to LDS then one TDM tensor_store_from_lds per block.
// ---------------------------------------------------------------------------
#define GEMM_AP 72    // LDS pitch (bf16) for A tile (64 K + pad)
#define GEMM_BP 136   // LDS pitch (bf16) for B tile (128 N + pad)

#if __has_builtin(__builtin_amdgcn_tensor_store_from_lds)
#define HAVE_TDM_STORE 1
#else
#define HAVE_TDM_STORE 0
#endif

template<int ACT, bool TRANSB, bool RESID, bool HAS_BIAS>
__global__ __launch_bounds__(256)
void gemm_kernel(const float* __restrict__ A, const float* __restrict__ W,
                 const float* __restrict__ bias, const float* __restrict__ resid,
                 float* __restrict__ C, int M, int N, int K) {
    const int BK = 64;
    __shared__ __align__(16) unsigned char smem[128 * 128 * 4];   // 64 KB
    unsigned short* As = (unsigned short*)smem;                       // 18.4KB
    unsigned short* Bs = (unsigned short*)(smem + 128 * GEMM_AP * 2); // 17.4KB
    float* Cs = (float*)smem;                                         // epilogue stage

    const int tid  = threadIdx.x;
    const int lane = tid & 31;
    const int wave = tid >> 5;
    const int wm = (wave & 3) * 32;
    const int wn = (wave >> 2) * 64;
    const int rowBase = blockIdx.y * 128;
    const int colBase = blockIdx.x * 128;

    v8f acc[2][4];
#pragma unroll
    for (int i = 0; i < 2; ++i)
#pragma unroll
        for (int j = 0; j < 4; ++j) {
            v8f z = {0.f, 0.f, 0.f, 0.f, 0.f, 0.f, 0.f, 0.f};
            acc[i][j] = z;
        }

    const int am = tid >> 1;
    const int ak = (tid & 1) * 32;

    for (int k0 = 0; k0 < K; k0 += BK) {
        {   // stage A (128x64) fp32->bf16
            const float* src = A + (size_t)(rowBase + am) * K + k0 + ak;
            unsigned short* dst = &As[am * GEMM_AP + ak];
            if (k0 + BK < K) __builtin_prefetch(src + BK, 0, 3);
#pragma unroll
            for (int i = 0; i < 32; i += 4) {
                float4 v = *(const float4*)(src + i);
                dst[i + 0] = f32_to_bf16_rne(v.x);
                dst[i + 1] = f32_to_bf16_rne(v.y);
                dst[i + 2] = f32_to_bf16_rne(v.z);
                dst[i + 3] = f32_to_bf16_rne(v.w);
            }
        }
        if (!TRANSB) {   // stage B (64x128)
            const int bk_ = tid >> 2;
            const int bn_ = (tid & 3) * 32;
            const float* src = W + (size_t)(k0 + bk_) * N + colBase + bn_;
            unsigned short* dst = &Bs[bk_ * GEMM_BP + bn_];
            if (k0 + BK < K) __builtin_prefetch(src + (size_t)BK * N, 0, 3);
#pragma unroll
            for (int i = 0; i < 32; i += 4) {
                float4 v = *(const float4*)(src + i);
                dst[i + 0] = f32_to_bf16_rne(v.x);
                dst[i + 1] = f32_to_bf16_rne(v.y);
                dst[i + 2] = f32_to_bf16_rne(v.z);
                dst[i + 3] = f32_to_bf16_rne(v.w);
            }
        } else {         // B[k][n] = W[n*K+k] (LM head)
            const int bn_ = tid >> 1;
            const int bkT = (tid & 1) * 32;
            const int nG  = colBase + bn_;
            if (nG < N) {
                const float* src = W + (size_t)nG * K + k0 + bkT;
                if (k0 + BK < K) __builtin_prefetch(src + BK, 0, 3);
#pragma unroll
                for (int i = 0; i < 32; i += 4) {
                    float4 v = *(const float4*)(src + i);
                    Bs[(bkT + i + 0) * GEMM_BP + bn_] = f32_to_bf16_rne(v.x);
                    Bs[(bkT + i + 1) * GEMM_BP + bn_] = f32_to_bf16_rne(v.y);
                    Bs[(bkT + i + 2) * GEMM_BP + bn_] = f32_to_bf16_rne(v.z);
                    Bs[(bkT + i + 3) * GEMM_BP + bn_] = f32_to_bf16_rne(v.w);
                }
            } else {
#pragma unroll
                for (int i = 0; i < 32; ++i) Bs[(bkT + i) * GEMM_BP + bn_] = 0;
            }
        }
        __syncthreads();

        const int mrow = lane & 15;
        const int kb   = (lane >> 4) * 8;
#pragma unroll
        for (int ks = 0; ks < 2; ++ks) {
            Frag a[2], b[4];
#pragma unroll
            for (int mi = 0; mi < 2; ++mi) {
                const unsigned short* base =
                    &As[(wm + 16 * mi + mrow) * GEMM_AP + ks * 32];
#pragma unroll
                for (int j = 0; j < 8; ++j)
                    a[mi].u[j] = *(const unsigned*)(base + a_kcol(j, kb));
            }
#pragma unroll
            for (int ni = 0; ni < 4; ++ni) {
                const unsigned short* base =
                    &Bs[(ks * 32) * GEMM_BP + wn + 16 * ni + mrow];
#pragma unroll
                for (int j = 0; j < 8; ++j) {
                    const int kk = a_kcol(j, kb);
                    b[ni].u[j] = (unsigned)base[kk * GEMM_BP] | ((unsigned)base[(kk + 1) * GEMM_BP] << 16);
                }
            }
#pragma unroll
            for (int mi = 0; mi < 2; ++mi)
#pragma unroll
                for (int ni = 0; ni < 4; ++ni)
                    acc[mi][ni] = __builtin_amdgcn_wmma_f32_16x16x32_bf16(
                        false, a[mi].v, false, b[ni].v,
                        (short)0, acc[mi][ni], false, false);
        }
        __syncthreads();
    }

    const int tileW = (N - colBase < 128) ? (N - colBase) : 128;
#if HAVE_TDM_STORE
#pragma unroll
    for (int mi = 0; mi < 2; ++mi) {
#pragma unroll
        for (int ni = 0; ni < 4; ++ni) {
            const int col = wn + 16 * ni + (lane & 15);
            const bool ok = col < tileW;
            float bv = 0.f;
            if (HAS_BIAS && ok) bv = bias[colBase + col];
#pragma unroll
            for (int r = 0; r < 8; ++r) {
                if (ok) {
                    const int row = wm + 16 * mi + r + 8 * (lane >> 4);
                    float v = acc[mi][ni][r];
                    if (HAS_BIAS) v += bv;
                    if (ACT == 1)
                        v = 0.5f * v * (1.0f + erff(v * 0.70710678118654752f));
                    if (RESID)
                        v += resid[(size_t)(rowBase + row) * N + colBase + col];
                    Cs[row * tileW + col] = v;
                }
            }
        }
    }
    __syncthreads();
    if (wave == 0) {
        unsigned long long ga =
            (unsigned long long)(size_t)(C + (size_t)rowBase * N + colBase);
        unsigned lds_off = (unsigned)(size_t)Cs;
        v4u g0 = { 1u, lds_off,
                   (unsigned)(ga & 0xFFFFFFFFu),
                   (unsigned)((ga >> 32) & 0x01FFFFFFu) | (2u << 30) };
        const unsigned dim0 = (unsigned)tileW;
        const unsigned dim1 = 128u;
        v8i g1 = { (int)(2u << 16),
                   (int)((dim0 & 0xFFFFu) << 16),
                   (int)((dim0 >> 16) | ((dim1 & 0xFFFFu) << 16)),
                   (int)((dim1 >> 16) | ((unsigned)tileW << 16)),
                   (int)128u,
                   (int)(unsigned)N,
                   0, 0 };
        v4i g2 = { 1, 1, 0, 0 };
        v4i g3 = { 0, 0, 0, 0 };
#if defined(__clang_major__) && (__clang_major__ >= 23)
        v8i g2b = { g2.x, g2.y, g2.z, g2.w, 0, 0, 0, 0 };
        __builtin_amdgcn_tensor_store_from_lds(g0, g1, g2, g3, g2b, 0);
#else
        __builtin_amdgcn_tensor_store_from_lds(g0, g1, g2, g3, 0);
#endif
        __builtin_amdgcn_s_wait_tensorcnt(0);
    }
#else
#pragma unroll
    for (int mi = 0; mi < 2; ++mi) {
#pragma unroll
        for (int ni = 0; ni < 4; ++ni) {
            const int col = wn + 16 * ni + (lane & 15);
            const bool ok = col < tileW;
            float bv = 0.f;
            if (HAS_BIAS && ok) bv = bias[colBase + col];
#pragma unroll
            for (int r = 0; r < 8; ++r) {
                if (ok) {
                    const int row = wm + 16 * mi + r + 8 * (lane >> 4);
                    float v = acc[mi][ni][r];
                    if (HAS_BIAS) v += bv;
                    if (ACT == 1)
                        v = 0.5f * v * (1.0f + erff(v * 0.70710678118654752f));
                    size_t off = (size_t)(rowBase + row) * N + colBase + col;
                    if (RESID) v += resid[off];
                    C[off] = v;
                }
            }
        }
    }
#endif
}

// ---------------------------------------------------------------------------
// Flash attention with WMMA (bf16 inputs, f32 accumulate, online softmax).
// Grid: (T/128, B*H). 8 waves; wave owns 16 query rows. Per 64-key block:
//   S = Q K^T    (8 wmma)  -> scale, causal mask, online softmax
//   O = a*O + P V (8 wmma), P round-tripped through per-wave LDS tile to
//   convert C-layout -> A-fragment layout (wave-local DS ordering, no barrier)
// ---------------------------------------------------------------------------
#define FA_P 72   // LDS pitch (bf16 elems)

__global__ __launch_bounds__(256)
void flash_attn_kernel(const float* __restrict__ qkv, float* __restrict__ o) {
    __shared__ unsigned short Qs[128 * FA_P];     // 18.4KB  [q][d]
    __shared__ unsigned short Kt[64 * FA_P];      //  9.2KB  [d][key]   (transposed)
    __shared__ unsigned short Vs[64 * FA_P];      //  9.2KB  [key][d]
    __shared__ unsigned short Ps[8 * 16 * FA_P];  // 18.4KB  per-wave P tile [q][key]

    const int tid  = threadIdx.x;
    const int lane = tid & 31;
    const int wave = tid >> 5;
    const int qb = blockIdx.x;           // query block (0..7)
    const int bh = blockIdx.y;           // 0..63
    const int h  = bh & (NH - 1);
    const int b  = bh >> 4;
    const int qBase = qb * 128;
    const size_t rs = 3 * (size_t)ND;
    const float* base = qkv + (size_t)(b * NT) * rs + h * NHD;

    // ---- stage Q tile (128x64) fp32->bf16 ----
    {
        const int q  = tid >> 1;
        const int d0 = (tid & 1) * 32;
        const float* src = base + (size_t)(qBase + q) * rs + d0;
        unsigned short* dst = &Qs[q * FA_P + d0];
#pragma unroll
        for (int i = 0; i < 32; i += 4) {
            float4 v = *(const float4*)(src + i);
            dst[i + 0] = f32_to_bf16_rne(v.x);
            dst[i + 1] = f32_to_bf16_rne(v.y);
            dst[i + 2] = f32_to_bf16_rne(v.z);
            dst[i + 3] = f32_to_bf16_rne(v.w);
        }
    }
    __syncthreads();

    // ---- Q A-fragments (fixed for whole kernel): 2 K-chunks of 32 ----
    const int mrow = lane & 15;
    const int kb8  = (lane >> 4) * 8;
    const int hi8  = (lane >> 4) * 8;
    Frag aq[2];
#pragma unroll
    for (int ks = 0; ks < 2; ++ks) {
        const unsigned short* qbase = &Qs[(wave * 16 + mrow) * FA_P + ks * 32];
#pragma unroll
        for (int j = 0; j < 8; ++j)
            aq[ks].u[j] = *(const unsigned*)(qbase + a_kcol(j, kb8));
    }

    float m_row[8], l_row[8];
    v8f acc_o[4];
#pragma unroll
    for (int r = 0; r < 8; ++r) { m_row[r] = -INFINITY; l_row[r] = 0.f; }
#pragma unroll
    for (int ni = 0; ni < 4; ++ni) {
        v8f z = {0.f, 0.f, 0.f, 0.f, 0.f, 0.f, 0.f, 0.f};
        acc_o[ni] = z;
    }

    unsigned short* pw = &Ps[wave * 16 * FA_P];

    for (int kb0 = 0; kb0 < qBase + 128; kb0 += 64) {
        // ---- stage K^T (d x key) and V (key x d) for this key block ----
        {
            const int key = tid >> 2;
            const int d0  = (tid & 3) * 16;
            const float* ksrc = base + (size_t)(kb0 + key) * rs + ND + d0;
            const float* vsrc = base + (size_t)(kb0 + key) * rs + 2 * ND + d0;
#pragma unroll
            for (int i = 0; i < 16; i += 4) {
                float4 kv = *(const float4*)(ksrc + i);
                Kt[(d0 + i + 0) * FA_P + key] = f32_to_bf16_rne(kv.x);
                Kt[(d0 + i + 1) * FA_P + key] = f32_to_bf16_rne(kv.y);
                Kt[(d0 + i + 2) * FA_P + key] = f32_to_bf16_rne(kv.z);
                Kt[(d0 + i + 3) * FA_P + key] = f32_to_bf16_rne(kv.w);
                float4 vv = *(const float4*)(vsrc + i);
                unsigned short* vd = &Vs[key * FA_P + d0 + i];
                vd[0] = f32_to_bf16_rne(vv.x);
                vd[1] = f32_to_bf16_rne(vv.y);
                vd[2] = f32_to_bf16_rne(vv.z);
                vd[3] = f32_to_bf16_rne(vv.w);
            }
        }
        __syncthreads();

        // ---- S = Q K^T : 4 tiles (keys) x 2 K-steps (d) ----
        v8f s_acc[4];
#pragma unroll
        for (int ni = 0; ni < 4; ++ni) {
            v8f z = {0.f, 0.f, 0.f, 0.f, 0.f, 0.f, 0.f, 0.f};
            s_acc[ni] = z;
        }
#pragma unroll
        for (int ks = 0; ks < 2; ++ks) {
            Frag bk[4];
#pragma unroll
            for (int ni = 0; ni < 4; ++ni) {
                const unsigned short* kb_ = &Kt[(ks * 32) * FA_P + ni * 16 + mrow];
#pragma unroll
                for (int j = 0; j < 8; ++j) {
                    const int kk = a_kcol(j, kb8);
                    bk[ni].u[j] = (unsigned)kb_[kk * FA_P] | ((unsigned)kb_[(kk + 1) * FA_P] << 16);
                }
            }
#pragma unroll
            for (int ni = 0; ni < 4; ++ni)
                s_acc[ni] = __builtin_amdgcn_wmma_f32_16x16x32_bf16(
                    false, aq[ks].v, false, bk[ni].v,
                    (short)0, s_acc[ni], false, false);
        }

        // ---- scale + causal mask + online softmax update ----
        const bool needMask = (kb0 + 63) > (qBase + wave * 16);  // uniform per wave
        float alpha[8];
#pragma unroll
        for (int r = 0; r < 8; ++r) {
            const int rowG = qBase + wave * 16 + r + hi8;
            float rm = -INFINITY;
#pragma unroll
            for (int ni = 0; ni < 4; ++ni) {
                float sv = s_acc[ni][r] * 0.125f;      // 1/sqrt(64)
                if (needMask) {
                    const int colG = kb0 + ni * 16 + (lane & 15);
                    sv = (colG <= rowG) ? sv : -INFINITY;
                }
                s_acc[ni][r] = sv;
                rm = fmaxf(rm, sv);
            }
            rm = group16_max(rm);
            const float mn = fmaxf(m_row[r], rm);
            alpha[r] = __expf(m_row[r] - mn);
            float psum = 0.f;
#pragma unroll
            for (int ni = 0; ni < 4; ++ni) {
                float p = __expf(s_acc[ni][r] - mn);
                s_acc[ni][r] = p;
                psum += p;
            }
            l_row[r] = l_row[r] * alpha[r] + group16_sum(psum);
            m_row[r] = mn;
        }

        // ---- P -> per-wave LDS tile (C-layout -> row-major bf16) ----
#pragma unroll
        for (int ni = 0; ni < 4; ++ni) {
            const int c = ni * 16 + (lane & 15);
#pragma unroll
            for (int r = 0; r < 8; ++r)
                pw[(r + hi8) * FA_P + c] = f32_to_bf16_rne(s_acc[ni][r]);
        }
        // rescale running output
#pragma unroll
        for (int ni = 0; ni < 4; ++ni)
#pragma unroll
            for (int r = 0; r < 8; ++r)
                acc_o[ni][r] *= alpha[r];

        // ---- O += P V : A-frags from Ps (wave-local), B-frags from Vs ----
#pragma unroll
        for (int ks = 0; ks < 2; ++ks) {
            Frag ap;
            const unsigned short* pb = &pw[mrow * FA_P + ks * 32];
#pragma unroll
            for (int j = 0; j < 8; ++j)
                ap.u[j] = *(const unsigned*)(pb + a_kcol(j, kb8));
            Frag bv[4];
#pragma unroll
            for (int ni = 0; ni < 4; ++ni) {
                const unsigned short* vb = &Vs[(ks * 32) * FA_P + ni * 16 + mrow];
#pragma unroll
                for (int j = 0; j < 8; ++j) {
                    const int kk = a_kcol(j, kb8);
                    bv[ni].u[j] = (unsigned)vb[kk * FA_P] | ((unsigned)vb[(kk + 1) * FA_P] << 16);
                }
            }
#pragma unroll
            for (int ni = 0; ni < 4; ++ni)
                acc_o[ni] = __builtin_amdgcn_wmma_f32_16x16x32_bf16(
                    false, ap.v, false, bv[ni].v,
                    (short)0, acc_o[ni], false, false);
        }
        __syncthreads();   // before next key block overwrites Kt/Vs
    }

    // ---- normalize and write O ----
#pragma unroll
    for (int ni = 0; ni < 4; ++ni) {
        const int c = ni * 16 + (lane & 15);
#pragma unroll
        for (int r = 0; r < 8; ++r) {
            const int row = qBase + wave * 16 + r + hi8;
            o[(size_t)(b * NT + row) * ND + h * NHD + c] = acc_o[ni][r] / l_row[r];
        }
    }
}

// ---------------------------------------------------------------------------
// Orchestration
// ---------------------------------------------------------------------------
extern "C" void kernel_launch(void* const* d_in, const int* in_sizes, int n_in,
                              void* d_out, int out_size, void* d_ws, size_t ws_size,
                              hipStream_t stream) {
    (void)in_sizes; (void)n_in; (void)out_size; (void)ws_size;
    const int*   idx   = (const int*)  d_in[0];
    const float* wte   = (const float*)d_in[1];
    const float* wpe   = (const float*)d_in[2];
    const float* ln1_g = (const float*)d_in[3];
    const float* ln1_b = (const float*)d_in[4];
    const float* ln2_g = (const float*)d_in[5];
    const float* ln2_b = (const float*)d_in[6];
    const float* qkv_w = (const float*)d_in[7];
    const float* proj_w= (const float*)d_in[8];
    const float* ff1_w = (const float*)d_in[9];
    const float* ff1_b = (const float*)d_in[10];
    const float* ff2_w = (const float*)d_in[11];
    const float* ff2_b = (const float*)d_in[12];
    const float* lnf_g = (const float*)d_in[13];
    const float* lnf_b = (const float*)d_in[14];
    float* out = (float*)d_out;

    float* x   = (float*)d_ws;                    // [4096,1024]
    float* h   = x   + (size_t)MTOT * ND;         // [4096,1024]
    float* qkv = h   + (size_t)MTOT * ND;         // [4096,3072]
    float* o   = qkv + (size_t)MTOT * 3 * ND;     // [4096,1024]
    float* ff  = o   + (size_t)MTOT * ND;         // [4096,4096]

    embed_kernel<<<(MTOT * ND) / 256, 256, 0, stream>>>(idx, wte, wpe, x);

    for (int l = 0; l < NL; ++l) {
        // reference quirk: ln2 before attention, ln1 before FFN
        ln_kernel<<<MTOT, 256, 0, stream>>>(x, ln2_g + l * ND, ln2_b + l * ND, h);
        gemm_kernel<0, false, false, false>
            <<<dim3(3 * ND / 128, MTOT / 128), 256, 0, stream>>>(
            h, qkv_w + (size_t)l * ND * 3 * ND, nullptr, nullptr, qkv, MTOT, 3 * ND, ND);
        flash_attn_kernel<<<dim3(NT / 128, NB * NH), 256, 0, stream>>>(qkv, o);
        gemm_kernel<0, false, true, false>
            <<<dim3(ND / 128, MTOT / 128), 256, 0, stream>>>(
            o, proj_w + (size_t)l * ND * ND, nullptr, x, x, MTOT, ND, ND);
        ln_kernel<<<MTOT, 256, 0, stream>>>(x, ln1_g + l * ND, ln1_b + l * ND, h);
        gemm_kernel<1, false, false, true>
            <<<dim3(4 * ND / 128, MTOT / 128), 256, 0, stream>>>(
            h, ff1_w + (size_t)l * ND * 4 * ND, ff1_b + (size_t)l * 4 * ND, nullptr,
            ff, MTOT, 4 * ND, ND);
        gemm_kernel<0, false, true, true>
            <<<dim3(ND / 128, MTOT / 128), 256, 0, stream>>>(
            ff, ff2_w + (size_t)l * 4 * ND * ND, ff2_b + (size_t)l * ND, x, x,
            MTOT, ND, 4 * ND);
    }

    ln_kernel<<<MTOT, 256, 0, stream>>>(x, lnf_g, lnf_b, h);
    // tied LM head: logits = h @ wte^T
    gemm_kernel<0, true, false, false>
        <<<dim3((NV + 127) / 128, MTOT / 128), 256, 0, stream>>>(
        h, wte, nullptr, nullptr, out, MTOT, NV, ND);
}
